// NanoAttention_60653528154672
// MI455X (gfx1250) — matrix operand order repaired and checked by reading, not compiled, not run on hardware
//
#include <hip/hip_runtime.h>
#include <hip/hip_bf16.h>

// ---------------------------------------------------------------------------
// NanoAttention forward for gfx1250 (CDNA5): bf16 WMMA GEMMs + flash attention
// GEMM B-panels staged into LDS by the Tensor Data Mover (TDM), double-buffered.
// ---------------------------------------------------------------------------

typedef __bf16 bf16_t;
typedef __attribute__((ext_vector_type(16))) __bf16 bf16x16;
typedef __attribute__((ext_vector_type(8)))  __bf16 bf16x8;
typedef __attribute__((ext_vector_type(4)))  __bf16 bf16x4;
typedef __attribute__((ext_vector_type(8)))  float  f32x8;
typedef __attribute__((ext_vector_type(4)))  float  f32x4;
typedef __attribute__((ext_vector_type(4)))  unsigned int u32x4;
typedef __attribute__((ext_vector_type(8)))  unsigned int u32x8;

static constexpr int TSEQ = 2048;
static constexpr int CEMB = 1024;
static constexpr int NH   = 16;
static constexpr int HD   = 64;
static constexpr int NB   = 4;    // batch

// LDS row pitch for TDM-staged B tiles: 32 bf16 (64 B = 16 dwords of data)
// + 4 dwords of TDM pad = 80 B.  Keeps ds_load_b128 16B-aligned and makes the
// 16 per-half-wave column reads bank-conflict free (20 dwords/row, gcd trick).
static constexpr int BPITCH = 40;   // elements (80 bytes)

// ---------------- WMMA helpers ----------------------------------------------

__device__ __forceinline__ f32x8 wmma_bf16f32(bf16x16 a, bf16x16 b, f32x8 c) {
  // D = A(16x32 bf16) * B(32x16 bf16) + C(16x16 f32)
  return __builtin_amdgcn_wmma_f32_16x16x32_bf16(
      /*neg_a=*/false, a, /*neg_b=*/false, b,
      /*c_mod=*/(short)0, c, /*reuse_a=*/false, /*reuse_b=*/false);
}

// A-fragment: 16x32 bf16 tile, row-major source with leading dim `ld`.
__device__ __forceinline__ bf16x16 frag_a_load(const bf16_t* base, int ld) {
  const int lane = threadIdx.x & 31;
  const int hw = lane >> 4;
  const int r  = lane & 15;
  const bf16_t* p0 = base + (size_t)r * ld + hw * 16;
  const bf16_t* p1 = p0 + 8;
  bf16x8 lo = *reinterpret_cast<const bf16x8*>(p0);
  bf16x8 hi = *reinterpret_cast<const bf16x8*>(p1);
  bf16x16 f;
#pragma unroll
  for (int e = 0; e < 8; ++e) { f[e] = lo[e]; f[e + 8] = hi[e]; }
  return f;
}

// B-fragment: 32x16 (K x N) where the SOURCE is stored [N][K] row-major.
__device__ __forceinline__ bf16x16 frag_b_load_nk(const bf16_t* base, int ld) {
  const int lane = threadIdx.x & 31;
  const int hw = lane >> 4;
  const int n  = lane & 15;
  return *reinterpret_cast<const bf16x16*>(base + (size_t)n * ld + hw * 16);
}

// ---------------- Tensor Data Mover: 2D tile -> LDS --------------------------
// Builds a D# (groups 0/1) per CDNA5 ISA and
// issues tensor_load_to_lds.  2D tile (VADDR2/3 omitted), data_size = 2 bytes,
// LDS padding: 4 dwords inserted every 16 dwords (pad_interval=3, pad_amount=3).
__device__ __forceinline__ void tdm_load_tile_2d(unsigned lds_off, const bf16_t* gaddr,
                                                 unsigned tensor_d0, unsigned tensor_d1,
                                                 unsigned tile_d0, unsigned tile_d1,
                                                 unsigned stride_d0) {
  const unsigned long long ga = (unsigned long long)(size_t)gaddr;
  u32x4 g0;
  g0[0] = 1u;                                              // count=1 (valid user D#)
  g0[1] = lds_off;                                         // lds_addr
  g0[2] = (unsigned)(ga & 0xFFFFFFFFu);                    // global_addr[31:0]
  g0[3] = (unsigned)((ga >> 32) & 0x01FFFFFFu) | 0x80000000u; // addr[56:32] | type=2
  u32x8 g1;
  g1[0] = (1u << 16)      // data_size = 1 -> 2 bytes
        | (1u << 20)      // pad_enable
        | (3u << 22)      // pad_interval: every 16 dwords
        | (3u << 25);     // pad_amount: 4 dwords
  g1[1] = (tensor_d0 & 0xFFFFu) << 16;                     // tensor_dim0[15:0]
  g1[2] = (tensor_d0 >> 16) | ((tensor_d1 & 0xFFFFu) << 16);
  g1[3] = (tensor_d1 >> 16) | (tile_d0 << 16);             // tile_dim0
  g1[4] = tile_d1;                                         // tile_dim1 (tile_dim2=0)
  g1[5] = stride_d0;                                       // tensor_dim0_stride[31:0]
  g1[6] = 0;                                               // stride hi / dim1_stride lo
  g1[7] = 0;
  asm volatile("tensor_load_to_lds %0, %1" :: "s"(g0), "s"(g1) : "memory");
}

// ---------------- f32 -> bf16 conversion -------------------------------------

__global__ void __launch_bounds__(256)
cvt_f32_to_bf16(const float* __restrict__ in, bf16_t* __restrict__ out, int n4) {
  int i = blockIdx.x * blockDim.x + threadIdx.x;
  if (i < n4) {
    f32x4 v = reinterpret_cast<const f32x4*>(in)[i];
    bf16x4 o;
#pragma unroll
    for (int e = 0; e < 4; ++e) o[e] = (bf16_t)v[e];
    reinterpret_cast<bf16x4*>(out)[i] = o;
  }
}

// ---------------- GEMM:  D[M,N] = A[M,K] * B[N,K]^T --------------------------
// block = 256 threads = 8 waves as 2(M) x 4(N); block tile 64 x 256.
// B panel (256 x 32 per K-step) is staged into LDS by the TDM, double-buffered;
// wave 0 issues the DMA and gates it with s_wait_tensorcnt + barrier.
// mode 0: epilogue scatters qkv -> q[BH,T,D], k[BH,T,D], vT[BH,D,T] (bf16)
// mode 1: epilogue writes f32 to `out` (the final projection)

__global__ void __launch_bounds__(256)
gemm_bf16_kernel(const bf16_t* __restrict__ A, const bf16_t* __restrict__ B,
                 int K, int N,
                 bf16_t* __restrict__ qout, bf16_t* __restrict__ kout,
                 bf16_t* __restrict__ vTout, float* __restrict__ out, int mode) {
  __shared__ __attribute__((aligned(128))) bf16_t btile[2][256 * BPITCH]; // 2 x 20KB

  const int w    = threadIdx.x >> 5;
  const int wm   = w >> 2;
  const int wn   = w & 3;
  const int lane = threadIdx.x & 31;
  const int hw   = lane >> 4;
  const int ln   = lane & 15;
  const int m0   = blockIdx.y * 64 + wm * 32;
  const int n0b  = blockIdx.x * 256;            // block's N origin

  // prologue: kick off DMA of the first B panel (wave 0 only; EXEC-independent)
  if (w == 0) {
    tdm_load_tile_2d((unsigned)(size_t)&btile[0][0], B + (size_t)n0b * K,
                     (unsigned)K, (unsigned)N, 32u, 256u, (unsigned)K);
  }

  f32x8 acc[2][4] = {};
  int buf = 0;
  for (int kk = 0; kk < K; kk += 32, buf ^= 1) {
    if (w == 0) __builtin_amdgcn_s_wait_tensorcnt(0);  // current panel landed
    __syncthreads();
    if (w == 0 && kk + 32 < K) {                       // overlap next DMA w/ math
      tdm_load_tile_2d((unsigned)(size_t)&btile[buf ^ 1][0],
                       B + (size_t)n0b * K + kk + 32,
                       (unsigned)K, (unsigned)N, 32u, 256u, (unsigned)K);
    }

    // issue ALL operand loads for this K-step up front, then run the WMMA chain
    bf16x16 a0 = frag_a_load(A + (size_t)m0 * K + kk, K);
    bf16x16 a1 = frag_a_load(A + (size_t)(m0 + 16) * K + kk, K);
    bf16x16 bj[4];
#pragma unroll
    for (int j = 0; j < 4; ++j) {
      const int r = wn * 64 + j * 16 + ln;             // column row-index in panel
      bj[j] = *reinterpret_cast<const bf16x16*>(
          &btile[buf][(size_t)r * BPITCH + hw * 16]);  // 2x ds_load_b128
    }
#pragma unroll
    for (int j = 0; j < 4; ++j) {
      acc[0][j] = wmma_bf16f32(a0, bj[j], acc[0][j]);
      acc[1][j] = wmma_bf16f32(a1, bj[j], acc[1][j]);
    }
    if (kk + 32 < K) {
      __builtin_prefetch(A + (size_t)m0 * K + kk + 32, 0, 1);
    }
  }

#pragma unroll
  for (int i = 0; i < 2; ++i)
#pragma unroll
    for (int j = 0; j < 4; ++j)
#pragma unroll
      for (int e = 0; e < 8; ++e) {
        const int row = m0 + i * 16 + e + 8 * hw;   // C/D layout: VGPR e, half hw
        const int col = n0b + wn * 64 + j * 16 + ln;
        const float v = acc[i][j][e];
        if (mode == 0) {
          const int b   = row >> 11;              // row = b*T + t
          const int t   = row & (TSEQ - 1);
          const int sec = col >> 10;              // 0:q 1:k 2:v
          const int cc  = col & (CEMB - 1);
          const int h   = cc >> 6;
          const int d   = cc & 63;
          const int bh  = b * NH + h;
          if (sec == 0)      qout [((size_t)bh * TSEQ + t) * HD + d] = (bf16_t)v;
          else if (sec == 1) kout [((size_t)bh * TSEQ + t) * HD + d] = (bf16_t)v;
          else               vTout[((size_t)bh * HD + d) * TSEQ + t] = (bf16_t)v;
        } else {
          out[(size_t)row * CEMB + col] = v;
        }
      }
}

// ---------------- Flash attention (causal) -----------------------------------
// grid = (T/128, B*H), 256 threads.  Each wave owns 16 query rows and streams
// KV in 32-column steps with online softmax.

__device__ __forceinline__ void rowreduce_max16(f32x8& v) {
#pragma unroll
  for (int off = 8; off; off >>= 1)
#pragma unroll
    for (int e = 0; e < 8; ++e) v[e] = fmaxf(v[e], __shfl_xor(v[e], off, 32));
}
__device__ __forceinline__ void rowreduce_sum16(f32x8& v) {
#pragma unroll
  for (int off = 8; off; off >>= 1)
#pragma unroll
    for (int e = 0; e < 8; ++e) v[e] += __shfl_xor(v[e], off, 32);
}

__global__ void __launch_bounds__(256)
flash_attn_kernel(const bf16_t* __restrict__ q, const bf16_t* __restrict__ k,
                  const bf16_t* __restrict__ vT, bf16_t* __restrict__ y) {
  __shared__ __attribute__((aligned(64))) bf16_t pbuf[8][16 * 32];

  const int w    = threadIdx.x >> 5;
  const int lane = threadIdx.x & 31;
  const int hw   = lane >> 4;
  const int ln   = lane & 15;
  const int bh   = blockIdx.y;
  const int b    = bh >> 4;
  const int h    = bh & 15;
  const int q0   = blockIdx.x * 128 + w * 16;

  const bf16_t* Q  = q  + (size_t)bh * TSEQ * HD;
  const bf16_t* Km = k  + (size_t)bh * TSEQ * HD;
  const bf16_t* Vt = vT + (size_t)bh * HD * TSEQ;

  const bf16x16 qa0 = frag_a_load(Q + (size_t)q0 * HD + 0,  HD);
  const bf16x16 qa1 = frag_a_load(Q + (size_t)q0 * HD + 32, HD);

  f32x8 o0 = {}, o1 = {}, o2 = {}, o3 = {};
  f32x8 m_run, l_run;
#pragma unroll
  for (int e = 0; e < 8; ++e) { m_run[e] = -1e30f; l_run[e] = 0.0f; }

  const float scale = 0.125f;   // 1/sqrt(64)
  const int kv_end = q0 + 16;   // causal bound (uniform per wave)

  for (int c0 = 0; c0 < kv_end; c0 += 32) {
    // preload the four K-tile fragments, then run the S = Q*K^T chain
    bf16x16 kb0 = frag_b_load_nk(Km + (size_t)c0 * HD + 0,  HD);
    bf16x16 kb1 = frag_b_load_nk(Km + (size_t)c0 * HD + 32, HD);
    bf16x16 kb2 = frag_b_load_nk(Km + (size_t)(c0 + 16) * HD + 0,  HD);
    bf16x16 kb3 = frag_b_load_nk(Km + (size_t)(c0 + 16) * HD + 32, HD);
    f32x8 s0 = {}, s1 = {};
    s0 = wmma_bf16f32(qa0, kb0, s0);
    s0 = wmma_bf16f32(qa1, kb1, s0);
    s1 = wmma_bf16f32(qa0, kb2, s1);
    s1 = wmma_bf16f32(qa1, kb3, s1);

    f32x8 mnew, alpha, psum;
#pragma unroll
    for (int e = 0; e < 8; ++e) {
      const int rr = q0 + e + 8 * hw;
      s0[e] = (c0 + ln      <= rr) ? s0[e] * scale : -1e30f;
      s1[e] = (c0 + 16 + ln <= rr) ? s1[e] * scale : -1e30f;
      mnew[e] = fmaxf(s0[e], s1[e]);
    }
    rowreduce_max16(mnew);
#pragma unroll
    for (int e = 0; e < 8; ++e) {
      mnew[e]  = fmaxf(mnew[e], m_run[e]);
      alpha[e] = __expf(m_run[e] - mnew[e]);
      s0[e]    = __expf(s0[e] - mnew[e]);
      s1[e]    = __expf(s1[e] - mnew[e]);
      psum[e]  = s0[e] + s1[e];
    }
    rowreduce_sum16(psum);
#pragma unroll
    for (int e = 0; e < 8; ++e) {
      l_run[e] = l_run[e] * alpha[e] + psum[e];
      m_run[e] = mnew[e];
      o0[e] *= alpha[e]; o1[e] *= alpha[e];
      o2[e] *= alpha[e]; o3[e] *= alpha[e];
      const int rr = e + 8 * hw;
      pbuf[w][rr * 32 + ln]      = (bf16_t)s0[e];
      pbuf[w][rr * 32 + 16 + ln] = (bf16_t)s1[e];
    }
    __builtin_amdgcn_wave_barrier();           // keep ds_store before ds_load
    const bf16x16 pa = frag_a_load(&pbuf[w][0], 32);  // re-read as A-fragment
    __builtin_amdgcn_wave_barrier();

    // preload the four V-tile fragments (V stored transposed), then P*V chain
    bf16x16 vb0 = frag_b_load_nk(Vt + (size_t)0  * TSEQ + c0, TSEQ);
    bf16x16 vb1 = frag_b_load_nk(Vt + (size_t)16 * TSEQ + c0, TSEQ);
    bf16x16 vb2 = frag_b_load_nk(Vt + (size_t)32 * TSEQ + c0, TSEQ);
    bf16x16 vb3 = frag_b_load_nk(Vt + (size_t)48 * TSEQ + c0, TSEQ);
    o0 = wmma_bf16f32(pa, vb0, o0);
    o1 = wmma_bf16f32(pa, vb1, o1);
    o2 = wmma_bf16f32(pa, vb2, o2);
    o3 = wmma_bf16f32(pa, vb3, o3);
  }

#pragma unroll
  for (int e = 0; e < 8; ++e) {
    const float inv = 1.0f / l_run[e];
    const int t = q0 + e + 8 * hw;
    bf16_t* yrow = y + ((size_t)b * TSEQ + t) * CEMB + h * HD;
    yrow[0  + ln] = (bf16_t)(o0[e] * inv);
    yrow[16 + ln] = (bf16_t)(o1[e] * inv);
    yrow[32 + ln] = (bf16_t)(o2[e] * inv);
    yrow[48 + ln] = (bf16_t)(o3[e] * inv);
  }
}

// ---------------- launch -----------------------------------------------------

extern "C" void kernel_launch(void* const* d_in, const int* in_sizes, int n_in,
                              void* d_out, int out_size, void* d_ws, size_t ws_size,
                              hipStream_t stream) {
  const float* x     = (const float*)d_in[0];   // [4,2048,1024]
  const float* Wqkv  = (const float*)d_in[1];   // [3072,1024]
  const float* Wproj = (const float*)d_in[2];   // [1024,1024]
  float* out = (float*)d_out;                   // [4,2048,1024] f32

  const size_t nX   = (size_t)NB * TSEQ * CEMB;      // 8388608
  const size_t nWq  = (size_t)3 * CEMB * CEMB;       // 3145728
  const size_t nWp  = (size_t)CEMB * CEMB;           // 1048576
  const size_t nHed = (size_t)NB * NH * TSEQ * HD;   // 8388608

  bf16_t* ws = (bf16_t*)d_ws;
  bf16_t* xb     = ws;             ws += nX;
  bf16_t* wqkvb  = ws;             ws += nWq;
  bf16_t* wprojb = ws;             ws += nWp;
  bf16_t* qb     = ws;             ws += nHed;
  bf16_t* kb     = ws;             ws += nHed;
  bf16_t* vtb    = ws;             ws += nHed;
  bf16_t* yb     = ws;             ws += nX;

  // 1) convert inputs to bf16
  cvt_f32_to_bf16<<<(nX  / 4 + 255) / 256, 256, 0, stream>>>(x,     xb,     (int)(nX  / 4));
  cvt_f32_to_bf16<<<(nWq / 4 + 255) / 256, 256, 0, stream>>>(Wqkv,  wqkvb,  (int)(nWq / 4));
  cvt_f32_to_bf16<<<(nWp / 4 + 255) / 256, 256, 0, stream>>>(Wproj, wprojb, (int)(nWp / 4));

  // 2) QKV projection: [8192,1024] x [3072,1024]^T, scatter into q/k/vT
  gemm_bf16_kernel<<<dim3(3 * CEMB / 256, NB * TSEQ / 64), 256, 0, stream>>>(
      xb, wqkvb, CEMB, 3 * CEMB, qb, kb, vtb, nullptr, 0);

  // 3) causal flash attention per (B,H)
  flash_attn_kernel<<<dim3(TSEQ / 128, NB * NH), 256, 0, stream>>>(qb, kb, vtb, yb);

  // 4) output projection: [8192,1024] x [1024,1024]^T -> f32 out
  gemm_bf16_kernel<<<dim3(CEMB / 256, NB * TSEQ / 64), 256, 0, stream>>>(
      yb, wprojb, CEMB, CEMB, nullptr, nullptr, nullptr, out, 1);
}
